// Decode_11793980195167
// MI455X (gfx1250) — matrix operand order repaired and checked by reading, not compiled, not run on hardware
//
#include <hip/hip_runtime.h>
#include <hip/hip_bf16.h>
#include <math.h>

// Problem constants (match reference)
#define BB   128
#define LL   64
#define EMB  512
#define UU   1024
#define VV   50257
#define BN_EPS 0.001f

typedef __attribute__((ext_vector_type(16))) _Float16 v16h;
typedef __attribute__((ext_vector_type(8)))  _Float16 v8h;
typedef __attribute__((ext_vector_type(8)))  float    v8f;

__device__ __forceinline__ v16h cat8(v8h lo, v8h hi) {
  return __builtin_shufflevector(lo, hi, 0, 1, 2, 3, 4, 5, 6, 7,
                                         8, 9, 10, 11, 12, 13, 14, 15);
}

// ---------------------------------------------------------------------------
// WMMA GEMM:  C[m,n] = sum_k A[m,k] * W[n,k] + bias[n]
//   A : MxK f16 row-major (pre-converted activations; K contiguous)
//   W : NxK f32 row-major (streamed weights; converted in-register)
// One wave per 16x64 output tile: the A fragment is loaded once per k-step
// (two global_load_b128 of packed f16) and reused by 4 WMMAs. W rows are
// streamed with in-register v_cvt_pk_f16_f32 and prefetched two k-steps
// ahead (global_prefetch_b8). N edge handled by clamp + store guard.
// M multiple of 16, K multiple of 32 (true for all calls here).
// ---------------------------------------------------------------------------
__global__ __launch_bounds__(32) void wmma_gemm_a16(
    const _Float16* __restrict__ A, const float* __restrict__ W,
    const float* __restrict__ bias, float* __restrict__ C,
    int M, int N, int K)
{
  const int lane  = threadIdx.x & 31;
  const int half  = lane >> 4;          // K-group within fragment
  const int idx   = lane & 15;          // row (A) / column (W) within tile
  const int mbase = blockIdx.y * 16;
  const int nb0   = blockIdx.x * 64;

  const _Float16* __restrict__ aptr = A + (size_t)(mbase + idx) * K;

  const float* wp[4];
  int col[4];
#pragma unroll
  for (int t = 0; t < 4; ++t) {
    col[t] = nb0 + t * 16 + idx;
    int r = (col[t] < N) ? col[t] : (N - 1);  // clamp: garbage col never stored
    wp[t] = W + (size_t)r * K;
  }

  v8f acc[4] = {};
  for (int k0 = 0; k0 < K; k0 += 32) {
    // A fragment (ISA 16-bit A 16x32 layout), direct packed-f16 loads
    v8h alo = *(const v8h*)(aptr + k0 + half * 8);
    v8h ahi = *(const v8h*)(aptr + k0 + 16 + half * 8);
    v16h a  = cat8(alo, ahi);

#pragma unroll
    for (int t = 0; t < 4; ++t) {
      __builtin_prefetch(wp[t] + k0 + 64, 0, 1);   // global_prefetch_b8
      v16h b;
#pragma unroll
      for (int j = 0; j < 16; ++j)
        b[j] = (_Float16)wp[t][k0 + ((j & 8) << 1) + half * 8 + (j & 7)];
      acc[t] = __builtin_amdgcn_wmma_f32_16x16x32_f16(
          false, a, false, b, (short)0, acc[t], false, false);
    }
  }

  // C/D layout: lanes 0-15 rows 0..7, lanes 16-31 rows 8..15; col = idx
#pragma unroll
  for (int t = 0; t < 4; ++t) {
    if (col[t] < N) {
      const float bv = bias ? bias[col[t]] : 0.f;
#pragma unroll
      for (int v = 0; v < 8; ++v) {
        int m = mbase + half * 8 + v;
        C[(size_t)m * N + col[t]] = acc[t][v] + bv;
      }
    }
  }
}

// ---------------------------------------------------------------------------
// Fused attention score:
//   score[b,l] = Vattn_b + sum_u tanh(Uf[b,l,u] + Uattn_b[u] + Wh[b,u]) * Vattn_w[u]
// with Uf = features @ Uattn_w^T.  One wave per 16-row strip of (B*L).
// The 16x512 feature strip is converted to f16 into LDS once; the 64 n-tile
// passes then fetch A fragments with ds_load_b128 and B fragments from the
// pre-converted f16 Uattn_w.  Epilogue: tanh + Vattn dot via shfl_xor
// reduction across the 16 columns. L=64 is a multiple of 16 -> one b/strip.
// ---------------------------------------------------------------------------
__global__ __launch_bounds__(32) void attn_score_kernel(
    const float* __restrict__ features, const _Float16* __restrict__ Uw16,
    const float* __restrict__ Uattn_b,  const float* __restrict__ Wh,
    const float* __restrict__ Vattn_w,  const float* __restrict__ Vattn_b,
    float* __restrict__ score)
{
  __shared__ __align__(16) _Float16 As[16 * EMB];   // 16 KB

  const int lane  = threadIdx.x & 31;
  const int half  = lane >> 4;
  const int idx   = lane & 15;
  const int mbase = blockIdx.x * 16;        // row strip in (B*L)
  const int b     = mbase / LL;

  // stage the feature strip as f16: each lane converts half a row
  {
    const int row = lane >> 1;
    const int cb  = (lane & 1) * (EMB / 2);
    const float* __restrict__ src = features + (size_t)(mbase + row) * EMB + cb;
    _Float16* dst = As + row * EMB + cb;
    for (int j = 0; j < EMB / 2; ++j)
      dst[j] = (_Float16)src[j];
  }
  __syncthreads();

  float rowacc[8];
#pragma unroll
  for (int v = 0; v < 8; ++v) rowacc[v] = 0.f;

  for (int nt = 0; nt < UU / 16; ++nt) {
    const int nbase = nt * 16;
    const _Float16* __restrict__ wptr = Uw16 + (size_t)(nbase + idx) * EMB;

    v8f acc = {};
    for (int k0 = 0; k0 < EMB; k0 += 32) {
      v8h alo = *(const v8h*)(As + idx * EMB + k0 + half * 8);
      v8h ahi = *(const v8h*)(As + idx * EMB + k0 + 16 + half * 8);
      v16h a  = cat8(alo, ahi);
      v8h blo = *(const v8h*)(wptr + k0 + half * 8);
      v8h bhi = *(const v8h*)(wptr + k0 + 16 + half * 8);
      v16h bm = cat8(blo, bhi);
      acc = __builtin_amdgcn_wmma_f32_16x16x32_f16(
          false, a, false, bm, (short)0, acc, false, false);
    }

    const float vw = Vattn_w[nbase + idx];
    const float ub = Uattn_b[nbase + idx];
    const float wh = Wh[(size_t)b * UU + nbase + idx];
#pragma unroll
    for (int v = 0; v < 8; ++v) {
      float val = tanhf(acc[v] + ub + wh) * vw;
#pragma unroll
      for (int m = 1; m < 16; m <<= 1)       // reduce across the 16 columns
        val += __shfl_xor(val, m, 32);
      rowacc[v] += val;
    }
  }

  if (idx == 0) {
    const float vb = Vattn_b[0];
#pragma unroll
    for (int v = 0; v < 8; ++v) {
      int m = mbase + half * 8 + v;
      score[m] = rowacc[v] + vb;
    }
  }
}

// f32 -> f16 elementwise convert (for hidden and Uattn_w)
__global__ void f32_to_f16_kernel(const float* __restrict__ src,
                                  _Float16* __restrict__ dst, int n)
{
  int i = blockIdx.x * blockDim.x + threadIdx.x;
  if (i < n) dst[i] = (_Float16)src[i];
}

// ---------------------------------------------------------------------------
// Softmax over L per batch row + context vector ctx[b,e] = sum_l attn*feat.
// Also writes attn to the output tensor.
// ---------------------------------------------------------------------------
__global__ void softmax_ctx_kernel(
    const float* __restrict__ score, const float* __restrict__ feat,
    float* __restrict__ attn_out, float* __restrict__ ctx)
{
  const int b = blockIdx.x;
  __shared__ float sa[LL];
  const int t = threadIdx.x;

  if (t < LL) sa[t] = score[b * LL + t];
  __syncthreads();
  if (t == 0) {
    float mx = -1e30f;
    for (int l = 0; l < LL; ++l) mx = fmaxf(mx, sa[l]);
    float s = 0.f;
    for (int l = 0; l < LL; ++l) { float e = expf(sa[l] - mx); sa[l] = e; s += e; }
    float inv = 1.f / s;
    for (int l = 0; l < LL; ++l) sa[l] *= inv;
  }
  __syncthreads();
  if (t < LL) attn_out[b * LL + t] = sa[t];

  for (int e = t; e < EMB; e += blockDim.x) {
    float acc = 0.f;
    for (int l = 0; l < LL; ++l)
      acc += sa[l] * feat[((size_t)b * LL + l) * EMB + e];
    ctx[(size_t)b * EMB + e] = acc;
  }
}

// gin[b,:] = concat(ctx[b], emb_table[x[b]]) written directly as f16
__global__ void concat_kernel(
    const float* __restrict__ ctx, const float* __restrict__ emb_table,
    const int* __restrict__ x, _Float16* __restrict__ gin16)
{
  const int b   = blockIdx.x;
  const int row = x[b];
  for (int u = threadIdx.x; u < UU; u += blockDim.x) {
    float v = (u < EMB) ? ctx[(size_t)b * EMB + u]
                        : emb_table[(size_t)row * EMB + (u - EMB)];
    gin16[(size_t)b * UU + u] = (_Float16)v;
  }
}

__device__ __forceinline__ float sigmoidf_(float v) {
  return 1.f / (1.f + expf(-v));
}

// GRU gate math; h_new written f32 (output tensor) and f16 (next GEMM's A)
__global__ void gru_kernel(
    const float* __restrict__ gi, const float* __restrict__ gh,
    const float* __restrict__ h, float* __restrict__ h_new,
    _Float16* __restrict__ h_new16)
{
  const int b = blockIdx.x;
  for (int u = threadIdx.x; u < UU; u += blockDim.x) {
    const size_t o = (size_t)b * 3 * UU;
    float r = sigmoidf_(gi[o + u]          + gh[o + u]);
    float z = sigmoidf_(gi[o + UU + u]     + gh[o + UU + u]);
    float n = tanhf    (gi[o + 2 * UU + u] + r * gh[o + 2 * UU + u]);
    float hv = h[(size_t)b * UU + u];
    float hn = (1.f - z) * n + z * hv;
    h_new  [(size_t)b * UU + u] = hn;
    h_new16[(size_t)b * UU + u] = (_Float16)hn;
  }
}

// BatchNorm inference -> f16 (feeds the fc2 GEMM as A)
__global__ void bn_kernel(
    const float* __restrict__ y, const float* __restrict__ gamma,
    const float* __restrict__ beta, const float* __restrict__ mean,
    const float* __restrict__ var, _Float16* __restrict__ ybn16)
{
  const int b = blockIdx.x;
  for (int u = threadIdx.x; u < UU; u += blockDim.x) {
    float v = y[(size_t)b * UU + u];
    float o = (v - mean[u]) * rsqrtf(var[u] + BN_EPS) * gamma[u] + beta[u];
    ybn16[(size_t)b * UU + u] = (_Float16)o;
  }
}

// ---------------------------------------------------------------------------
extern "C" void kernel_launch(void* const* d_in, const int* in_sizes, int n_in,
                              void* d_out, int out_size, void* d_ws, size_t ws_size,
                              hipStream_t stream) {
  (void)in_sizes; (void)n_in; (void)out_size; (void)ws_size;

  const int*   x        = (const int*)  d_in[0];
  const float* features = (const float*)d_in[1];
  const float* hidden   = (const float*)d_in[2];   // (1,B,U) -> flat B*U
  const float* emb      = (const float*)d_in[3];
  const float* Uattn_w  = (const float*)d_in[4];
  const float* Uattn_b  = (const float*)d_in[5];
  const float* Wattn_w  = (const float*)d_in[6];
  const float* Wattn_b  = (const float*)d_in[7];
  const float* Vattn_w  = (const float*)d_in[8];
  const float* Vattn_b  = (const float*)d_in[9];
  const float* W_ih     = (const float*)d_in[10];
  const float* W_hh     = (const float*)d_in[11];
  const float* b_ih     = (const float*)d_in[12];
  const float* b_hh     = (const float*)d_in[13];
  const float* fc_w     = (const float*)d_in[14];
  const float* fc_b     = (const float*)d_in[15];
  const float* bn_gamma = (const float*)d_in[16];
  const float* bn_beta  = (const float*)d_in[17];
  const float* bn_mean  = (const float*)d_in[18];
  const float* bn_var   = (const float*)d_in[19];
  const float* fc2_w    = (const float*)d_in[20];
  const float* fc2_b    = (const float*)d_in[21];

  // output regions: (logits, h_new, attn) concatenated flat
  float* out_logits = (float*)d_out;                       // B*V
  float* out_hnew   = out_logits + (size_t)BB * VV;        // B*U
  float* out_attn   = out_hnew   + (size_t)BB * UU;        // B*L

  // workspace layout: f32 region then f16 region (all offsets 16B-aligned)
  float* ws      = (float*)d_ws;
  float* ws_Wh   = ws;                            // B*U
  float* ws_sc   = ws_Wh  + (size_t)BB * UU;      // B*L
  float* ws_ctx  = ws_sc  + (size_t)BB * LL;      // B*EMB
  float* ws_gi   = ws_ctx + (size_t)BB * EMB;     // B*3U
  float* ws_gh   = ws_gi  + (size_t)BB * 3 * UU;  // B*3U
  float* ws_y    = ws_gh  + (size_t)BB * 3 * UU;  // B*U
  _Float16* ws_uw16  = (_Float16*)(ws_y + (size_t)BB * UU);
  _Float16* ws_h16   = ws_uw16  + (size_t)UU * EMB;   // B*U
  _Float16* ws_gin16 = ws_h16   + (size_t)BB * UU;    // B*U
  _Float16* ws_hn16  = ws_gin16 + (size_t)BB * UU;    // B*U
  _Float16* ws_ybn16 = ws_hn16  + (size_t)BB * UU;    // B*U

  const dim3 wave(32);

  // 0) pre-convert reused operands to f16
  hipLaunchKernelGGL(f32_to_f16_kernel, dim3((BB * UU) / 256), dim3(256), 0, stream,
                     hidden, ws_h16, BB * UU);
  hipLaunchKernelGGL(f32_to_f16_kernel, dim3((UU * EMB) / 256), dim3(256), 0, stream,
                     Uattn_w, ws_uw16, UU * EMB);

  // 1) Wh = h @ Wattn_w^T + Wattn_b            (128 x 1024 x 1024)
  hipLaunchKernelGGL(wmma_gemm_a16, dim3(UU / 64, BB / 16), wave, 0, stream,
                     ws_h16, Wattn_w, Wattn_b, ws_Wh, BB, UU, UU);

  // 2) fused Uf GEMM + tanh + Vattn dot -> score   (8192 x 1024 x 512)
  hipLaunchKernelGGL(attn_score_kernel, dim3((BB * LL) / 16), wave, 0, stream,
                     features, ws_uw16, Uattn_b, ws_Wh, Vattn_w, Vattn_b, ws_sc);

  // 3) softmax over L + context; writes attn output
  hipLaunchKernelGGL(softmax_ctx_kernel, dim3(BB), dim3(256), 0, stream,
                     ws_sc, features, out_attn, ws_ctx);

  // 4) gin = concat(ctx, emb_table[x]) -> f16
  hipLaunchKernelGGL(concat_kernel, dim3(BB), dim3(256), 0, stream,
                     ws_ctx, emb, x, ws_gin16);

  // 5) gi = gin @ W_ih^T + b_ih                (128 x 3072 x 1024)
  hipLaunchKernelGGL(wmma_gemm_a16, dim3(3 * UU / 64, BB / 16), wave, 0, stream,
                     ws_gin16, W_ih, b_ih, ws_gi, BB, 3 * UU, UU);

  // 6) gh = h @ W_hh^T + b_hh                  (128 x 3072 x 1024)
  hipLaunchKernelGGL(wmma_gemm_a16, dim3(3 * UU / 64, BB / 16), wave, 0, stream,
                     ws_h16, W_hh, b_hh, ws_gh, BB, 3 * UU, UU);

  // 7) GRU gates -> h_new (f32 output + f16 copy)
  hipLaunchKernelGGL(gru_kernel, dim3(BB), dim3(256), 0, stream,
                     ws_gi, ws_gh, hidden, out_hnew, ws_hn16);

  // 8) y = h_new @ fc_w^T + fc_b               (128 x 1024 x 1024)
  hipLaunchKernelGGL(wmma_gemm_a16, dim3(UU / 64, BB / 16), wave, 0, stream,
                     ws_hn16, fc_w, fc_b, ws_y, BB, UU, UU);

  // 9) batch-norm -> f16
  hipLaunchKernelGGL(bn_kernel, dim3(BB), dim3(256), 0, stream,
                     ws_y, bn_gamma, bn_beta, bn_mean, bn_var, ws_ybn16);

  // 10) logits = ybn @ fc2_w^T + fc2_b         (128 x 50257 x 1024) — HBM-bound
  hipLaunchKernelGGL(wmma_gemm_a16, dim3((VV + 63) / 64, BB / 16), wave, 0, stream,
                     ws_ybn16, fc2_w, fc2_b, out_logits, BB, VV, UU);
}
